// RNN_BiGRU_9603546874334
// MI455X (gfx1250) — matrix operand order repaired and checked by reading, not compiled, not run on hardware
//
#include <hip/hip_runtime.h>
#include <hip/hip_bf16.h>
#include <cstdint>

typedef __attribute__((ext_vector_type(16))) __bf16    v16bf;
typedef __attribute__((ext_vector_type(8)))  float     v8f;
typedef __attribute__((ext_vector_type(4)))  uint32_t  u32x4;
typedef __attribute__((ext_vector_type(8)))  int       i32x8;
typedef __attribute__((ext_vector_type(4)))  int       i32x4;

#define T_LEN 1024
#define BATCH 64
#define HDIM  256
#define G3    768

#if defined(__has_builtin)
#if __has_builtin(__builtin_amdgcn_tensor_load_to_lds) && __has_builtin(__builtin_amdgcn_s_wait_tensorcnt)
#define USE_TDM 1
#endif
#endif

union Frag { v16bf v; uint32_t u[8]; };

__device__ __forceinline__ uint32_t f2bf_bits(float f) {
    uint32_t u = __float_as_uint(f);
    uint32_t r = u + 0x7FFFu + ((u >> 16) & 1u);   // round-to-nearest-even
    return r >> 16;
}

// fast gate nonlinearities: v_exp_f32 + v_rcp_f32, no libm special-casing
__device__ __forceinline__ float fast_sigmoid(float x) {
    return __builtin_amdgcn_rcpf(1.f + __expf(-x));
}
__device__ __forceinline__ float fast_tanh(float x) {
    return 2.f * __builtin_amdgcn_rcpf(1.f + __expf(-2.f * x)) - 1.f;
}

#if defined(USE_TDM)
// 1-D contiguous TDM copy: 24576 bytes (3072 x 8B units) global -> LDS.
// D# per CDNA5 ISA; 6-arg builtin form (clang-23).
__device__ __forceinline__ void tdm_load_slab(const void* gsrc, uint32_t lds_off) {
    uint64_t ga = (uint64_t)(uintptr_t)gsrc;
    u32x4 g0;
    g0[0] = 1u;                                               // count=1
    g0[1] = lds_off;                                          // lds_addr
    g0[2] = (uint32_t)ga;                                     // global_addr[31:0]
    g0[3] = (uint32_t)((ga >> 32) & 0x1FFFFFFu) | (2u << 30); // addr[56:32] | type=2
    i32x8 g1;
    g1[0] = (int)(3u << 16);                 // wg_mask=0, data_size=3 (8B)
    g1[1] = (int)((3072u & 0xFFFFu) << 16);  // tensor_dim0[15:0]
    g1[2] = 0;                               // tensor_dim0[31:16], tensor_dim1[15:0]
    g1[3] = (int)((3072u & 0xFFFFu) << 16);  // tile_dim0 = 3072
    g1[4] = 1;                               // tile_dim1 = 1, tile_dim2 = 0
    g1[5] = 3072;                            // tensor_dim0_stride
    g1[6] = 0;
    g1[7] = 0;
    i32x4 z4 = {0, 0, 0, 0};
    i32x8 z8 = {0, 0, 0, 0, 0, 0, 0, 0};
    __builtin_amdgcn_tensor_load_to_lds(g0, g1, z4, z4, z8, 0);
}
#endif

// ---------------------------------------------------------------------------
// Pack (768 x cols) fp32 W (used as B = W^T) into WMMA bf16 B-fragment layout.
// ---------------------------------------------------------------------------
__global__ void pack_bfrag(const float* __restrict__ W, uint32_t* __restrict__ dst,
                           int cols, int nkt) {
    int idx = blockIdx.x * blockDim.x + threadIdx.x;
    int total = 48 * nkt * 32 * 8;
    if (idx >= total) return;
    int v    = idx & 7;
    int lane = (idx >> 3) & 31;
    int kt   = (idx >> 8) % nkt;
    int nt   = (idx >> 8) / nkt;
    int n     = lane & 15;
    int kbase = (lane >> 4) * 8;
    int k     = kt * 32 + kbase + (v < 4 ? 2 * v : 16 + 2 * (v - 4));
    int row   = nt * 16 + n;
    uint32_t lo = f2bf_bits(W[(size_t)row * cols + k]);
    uint32_t hi = f2bf_bits(W[(size_t)row * cols + k + 1]);
    dst[idx] = lo | (hi << 16);
}

// ---------------------------------------------------------------------------
// xg = y0 @ W_ih1^T + b. One wave per 16x16 tile, K=512 (16 wmma k-steps),
// software-pipelined (double-buffered fragments).
// ---------------------------------------------------------------------------
__global__ void __launch_bounds__(256)
gemm_xg(const uint32_t* __restrict__ y0, const uint32_t* __restrict__ wpack,
        const float* __restrict__ bias, __hip_bfloat16* __restrict__ xg) {
    int gw   = (blockIdx.x * blockDim.x + threadIdx.x) >> 5;
    int lane = threadIdx.x & 31;
    int nt = gw % 48;
    int mt = gw / 48;
    int m     = lane & 15;
    int khalf = (lane >> 4) * 8;
    const uint32_t* arow = y0 + (size_t)(mt * 16 + m) * 256;

    Frag A[2], B[2];
    auto loadA = [&](int kt, Frag& f) {
#pragma unroll
        for (int q = 0; q < 8; ++q) {
            int kb = (q < 4 ? 2 * q : 16 + 2 * (q - 4)) + khalf;
            f.u[q] = arow[(kt * 32 + kb) >> 1];
        }
    };
    auto loadB = [&](int kt, Frag& f) {
        const uint32_t* bp = wpack + (((size_t)nt * 16 + kt) * 32 + lane) * 8;
        *(uint4*)&f.u[0] = *(const uint4*)bp;
        *(uint4*)&f.u[4] = *(const uint4*)(bp + 4);
    };

    v8f acc = {};
    loadA(0, A[0]);
    loadB(0, B[0]);
#pragma unroll
    for (int kt = 0; kt < 16; ++kt) {
        int cur = kt & 1;
        if (kt < 15) { loadA(kt + 1, A[cur ^ 1]); loadB(kt + 1, B[cur ^ 1]); }
        acc = __builtin_amdgcn_wmma_f32_16x16x32_bf16(false, A[cur].v, false, B[cur].v,
                                                      (short)0, acc, false, false);
    }

    int n    = lane & 15;
    int roff = (lane >> 4) * 8;
    float bv = bias[nt * 16 + n];
#pragma unroll
    for (int r = 0; r < 8; ++r) {
        int row = mt * 16 + roff + r;
        xg[(size_t)row * G3 + nt * 16 + n] = __float2bfloat16(acc[r] + bv);
    }
}

// ---------------------------------------------------------------------------
// One GRU layer, both directions. grid.x=8: dir=bx>>2, batch-quarter=bx&3.
// 16 waves; wave w owns column block w*16 and its r/z/n gate tiles, with the
// full W_hh slice register-resident (24 B-fragments = 192 VGPRs) across the
// 1024-step loop. h (16x256 bf16) in LDS. Layer-1 xg slabs staged to LDS by
// the Tensor Data Mover, double-buffered, waited with s_wait_tensorcnt.
// ---------------------------------------------------------------------------
__global__ void __launch_bounds__(512)
gru_layer(const float* __restrict__ x_in,
          const __hip_bfloat16* __restrict__ xg_f,
          const __hip_bfloat16* __restrict__ xg_b,
          const float* wih0_f, const float* wih0_b,
          const float* bih_f,  const float* bih_b,
          const float* bhh_f,  const float* bhh_b,
          const uint32_t* whh_f, const uint32_t* whh_b,
          __hip_bfloat16* __restrict__ y_out,
          float* __restrict__ hid_out,
          int layer0) {
    __shared__ uint32_t h_lds[16 * 128];                  // 16 x 256 bf16
#if defined(USE_TDM)
    __shared__ __hip_bfloat16 xg_lds[2][16 * G3];         // double-buffered xg slab
#endif

    int dir = blockIdx.x >> 2;
    int bs  = blockIdx.x & 3;
    int tid = threadIdx.x;
    int lane = tid & 31;
    int wv   = tid >> 5;

    const __hip_bfloat16* xg = dir ? xg_b : xg_f;
    const float*   wih0 = dir ? wih0_b : wih0_f;
    const float*   bih  = dir ? bih_b  : bih_f;
    const float*   bhh  = dir ? bhh_b  : bhh_f;
    const uint32_t* whh = dir ? whh_b  : whh_f;

    int n     = lane & 15;
    int c     = wv * 16 + n;
    int roff  = (lane >> 4) * 8;
    int m_a   = lane & 15;
    int khalf = (lane >> 4) * 8;
    int col_y = dir * HDIM + c;

    // ---- register-resident W_hh B-fragments: 3 gates x 8 k-steps ----
    Frag Bw[3][8];
#pragma unroll
    for (int g = 0; g < 3; ++g)
#pragma unroll
        for (int kt = 0; kt < 8; ++kt) {
            const uint32_t* p = whh + (((size_t)(wv + g * 16) * 8 + kt) * 32 + lane) * 8;
            *(uint4*)&Bw[g][kt].u[0] = *(const uint4*)p;
            *(uint4*)&Bw[g][kt].u[4] = *(const uint4*)(p + 4);
        }

    float bhh_r = bhh[c], bhh_z = bhh[c + 256], bhh_n = bhh[c + 512];
    float wx_r = 0.f, wx_z = 0.f, wx_n = 0.f, bx_r = 0.f, bx_z = 0.f, bx_n = 0.f;
    if (layer0) {
        wx_r = wih0[c]; wx_z = wih0[c + 256]; wx_n = wih0[c + 512];
        bx_r = bih[c];  bx_z = bih[c + 256];  bx_n = bih[c + 512];
    }

    for (int i = tid; i < 16 * 128; i += 512) h_lds[i] = 0;

#if defined(USE_TDM)
    size_t slab0 = ((size_t)(dir ? (T_LEN - 1) : 0) * BATCH + bs * 16) * G3;
    if (!layer0 && wv == 0) {
        tdm_load_slab(xg + slab0, (uint32_t)(uintptr_t)(void*)&xg_lds[0][0]);
        __builtin_amdgcn_s_wait_tensorcnt(0);
    }
#endif
    __syncthreads();

    __hip_bfloat16* h_bf = reinterpret_cast<__hip_bfloat16*>(h_lds);

    for (int s = 0; s < T_LEN; ++s) {
        int t = dir ? (T_LEN - 1 - s) : s;

#if defined(USE_TDM)
        if (!layer0 && wv == 0 && s + 1 < T_LEN) {
            int tn = dir ? (T_LEN - 2 - s) : (s + 1);
            size_t slab = ((size_t)tn * BATCH + bs * 16) * G3;
            tdm_load_slab(xg + slab, (uint32_t)(uintptr_t)(void*)&xg_lds[(s + 1) & 1][0]);
        }
#endif

        v8f a0 = {}, a1 = {}, a2 = {};
#pragma unroll
        for (int kt = 0; kt < 8; ++kt) {
            Frag A;
#pragma unroll
            for (int q = 0; q < 8; ++q) {
                int kb = (q < 4 ? 2 * q : 16 + 2 * (q - 4)) + khalf;
                A.u[q] = h_lds[m_a * 128 + ((kt * 32 + kb) >> 1)];
            }
            a0 = __builtin_amdgcn_wmma_f32_16x16x32_bf16(false, A.v, false, Bw[0][kt].v, (short)0, a0, false, false);
            a1 = __builtin_amdgcn_wmma_f32_16x16x32_bf16(false, A.v, false, Bw[1][kt].v, (short)0, a1, false, false);
            a2 = __builtin_amdgcn_wmma_f32_16x16x32_bf16(false, A.v, false, Bw[2][kt].v, (short)0, a2, false, false);
        }
        __syncthreads();   // all waves done reading old h

        size_t trow = (size_t)t * BATCH;
#pragma unroll
        for (int r = 0; r < 8; ++r) {
            int m = roff + r;
            int b = bs * 16 + m;
            float xr, xz, xn;
            if (layer0) {
                float xv = x_in[(size_t)b * T_LEN + t];
                xr = xv * wx_r + bx_r; xz = xv * wx_z + bx_z; xn = xv * wx_n + bx_n;
            } else {
#if defined(USE_TDM)
                const __hip_bfloat16* xrow = &xg_lds[s & 1][m * G3];
#else
                const __hip_bfloat16* xrow = xg + (trow + b) * G3;
#endif
                xr = __bfloat162float(xrow[c]);
                xz = __bfloat162float(xrow[c + 256]);
                xn = __bfloat162float(xrow[c + 512]);
            }
            float hr = a0[r] + bhh_r, hz = a1[r] + bhh_z, hn = a2[r] + bhh_n;
            float rg = fast_sigmoid(xr + hr);
            float zg = fast_sigmoid(xz + hz);
            float ng = fast_tanh(xn + rg * hn);
            float hprev = __bfloat162float(h_bf[m * HDIM + c]);
            float hnew  = (1.f - zg) * ng + zg * hprev;
            y_out[(trow + b) * 512 + col_y] = __float2bfloat16(hnew);
            h_bf[m * HDIM + c] = __float2bfloat16(hnew);
            if (s == T_LEN - 1)
                hid_out[(size_t)dir * (BATCH * HDIM) + (size_t)b * HDIM + c] = hnew;
        }
#if defined(USE_TDM)
        if (!layer0 && wv == 0)
            __builtin_amdgcn_s_wait_tensorcnt(0);   // next slab landed
#endif
        __syncthreads();   // new h (and next xg slab) visible to all
    }
}

// ---------------------------------------------------------------------------
// out[b][t][n] = sum_d y1[t*64+b][d] * cls_w[n][d] + cls_b[n], n < 8
// ---------------------------------------------------------------------------
__global__ void __launch_bounds__(256)
classifier(const __hip_bfloat16* __restrict__ y1, const float* __restrict__ cw,
           const float* __restrict__ cb, float* __restrict__ out) {
    __shared__ float cwl[8 * 512];
    int tid = threadIdx.x;
    for (int i = tid; i < 8 * 512; i += 256) cwl[i] = cw[i];
    __syncthreads();

    int gid = blockIdx.x * blockDim.x + tid;      // b*1024 + t
    int b = gid >> 10, t = gid & 1023;
    const __hip_bfloat16* row = y1 + ((size_t)t * BATCH + b) * 512;
    float acc[8];
#pragma unroll
    for (int nl = 0; nl < 8; ++nl) acc[nl] = cb[nl];
    for (int d = 0; d < 512; ++d) {
        float xv = __bfloat162float(row[d]);
#pragma unroll
        for (int nl = 0; nl < 8; ++nl) acc[nl] += xv * cwl[nl * 512 + d];
    }
#pragma unroll
    for (int nl = 0; nl < 8; ++nl) out[(size_t)gid * 8 + nl] = acc[nl];
}

// ---------------------------------------------------------------------------
extern "C" void kernel_launch(void* const* d_in, const int* in_sizes, int n_in,
                              void* d_out, int out_size, void* d_ws, size_t ws_size,
                              hipStream_t stream) {
    (void)in_sizes; (void)n_in; (void)out_size; (void)ws_size;
    const float* input    = (const float*)d_in[0];
    const float* wih_l0_f = (const float*)d_in[1];
    const float* whh_l0_f = (const float*)d_in[2];
    const float* bih_l0_f = (const float*)d_in[3];
    const float* bhh_l0_f = (const float*)d_in[4];
    const float* wih_l0_b = (const float*)d_in[5];
    const float* whh_l0_b = (const float*)d_in[6];
    const float* bih_l0_b = (const float*)d_in[7];
    const float* bhh_l0_b = (const float*)d_in[8];
    const float* wih_l1_f = (const float*)d_in[9];
    const float* whh_l1_f = (const float*)d_in[10];
    const float* bih_l1_f = (const float*)d_in[11];
    const float* bhh_l1_f = (const float*)d_in[12];
    const float* wih_l1_b = (const float*)d_in[13];
    const float* whh_l1_b = (const float*)d_in[14];
    const float* bih_l1_b = (const float*)d_in[15];
    const float* bhh_l1_b = (const float*)d_in[16];
    const float* cls_w    = (const float*)d_in[17];
    const float* cls_b    = (const float*)d_in[18];

    uint8_t* ws = (uint8_t*)d_ws;
    const size_t WHH_PACK_B  = (size_t)48 * 8 * 32 * 8 * 4;
    const size_t WIH1_PACK_B = (size_t)48 * 16 * 32 * 8 * 4;
    uint32_t* whh0f_p = (uint32_t*)(ws + 0 * WHH_PACK_B);
    uint32_t* whh0b_p = (uint32_t*)(ws + 1 * WHH_PACK_B);
    uint32_t* whh1f_p = (uint32_t*)(ws + 2 * WHH_PACK_B);
    uint32_t* whh1b_p = (uint32_t*)(ws + 3 * WHH_PACK_B);
    uint8_t*  base    = ws + 4 * WHH_PACK_B;
    uint32_t* wih1f_p = (uint32_t*)(base);
    uint32_t* wih1b_p = (uint32_t*)(base + WIH1_PACK_B);
    base += 2 * WIH1_PACK_B;
    const size_t Y_ELEMS = (size_t)T_LEN * BATCH * 512;
    __hip_bfloat16* y0 = (__hip_bfloat16*)base;  base += Y_ELEMS * 2;
    __hip_bfloat16* y1 = (__hip_bfloat16*)base;  base += Y_ELEMS * 2;
    const size_t XG_ELEMS = (size_t)T_LEN * BATCH * G3;
    __hip_bfloat16* xg_f = (__hip_bfloat16*)base;  base += XG_ELEMS * 2;
    __hip_bfloat16* xg_b = (__hip_bfloat16*)base;

    float* outp = (float*)d_out;
    float* hid  = outp + (size_t)BATCH * T_LEN * 8;

    pack_bfrag<<<384, 256, 0, stream>>>(whh_l0_f, whh0f_p, 256, 8);
    pack_bfrag<<<384, 256, 0, stream>>>(whh_l0_b, whh0b_p, 256, 8);
    pack_bfrag<<<384, 256, 0, stream>>>(whh_l1_f, whh1f_p, 256, 8);
    pack_bfrag<<<384, 256, 0, stream>>>(whh_l1_b, whh1b_p, 256, 8);
    pack_bfrag<<<768, 256, 0, stream>>>(wih_l1_f, wih1f_p, 512, 16);
    pack_bfrag<<<768, 256, 0, stream>>>(wih_l1_b, wih1b_p, 512, 16);

    gru_layer<<<8, 512, 0, stream>>>(input, nullptr, nullptr,
                                     wih_l0_f, wih_l0_b, bih_l0_f, bih_l0_b,
                                     bhh_l0_f, bhh_l0_b, whh0f_p, whh0b_p,
                                     y0, hid, 1);

    gemm_xg<<<24576, 256, 0, stream>>>((const uint32_t*)y0, wih1f_p, bih_l1_f, xg_f);
    gemm_xg<<<24576, 256, 0, stream>>>((const uint32_t*)y0, wih1b_p, bih_l1_b, xg_b);

    gru_layer<<<8, 512, 0, stream>>>(nullptr, xg_f, xg_b,
                                     nullptr, nullptr, bih_l1_f, bih_l1_b,
                                     bhh_l1_f, bhh_l1_b, whh1f_p, whh1b_p,
                                     y1, hid + 2 * BATCH * HDIM, 0);

    classifier<<<256, 256, 0, stream>>>(y1, cls_w, cls_b, outp);
}
